// TransformerBlock_22789096473365
// MI455X (gfx1250) — matrix operand order repaired and checked by reading, not compiled, not run on hardware
//
#include <hip/hip_runtime.h>

typedef __attribute__((ext_vector_type(16))) _Float16 v16h;
typedef __attribute__((ext_vector_type(8)))  _Float16 v8h;
typedef __attribute__((ext_vector_type(8)))  float    v8f;
typedef __attribute__((ext_vector_type(4)))  unsigned int v4u;
typedef __attribute__((ext_vector_type(8)))  int      v8i;
typedef __attribute__((ext_vector_type(4)))  int      v4i;

#ifndef __has_builtin
#define __has_builtin(x) 0
#endif
#if __has_builtin(__builtin_amdgcn_tensor_load_to_lds)
#define HAVE_TDM 1
#else
#define HAVE_TDM 0
#endif

enum { EPI_F16B = 0, EPI_F16RELU = 1, EPI_F16T = 2, EPI_F32B = 3, EPI_SIG = 4 };

// Staging geometry: 64x64 f16 panels, KT=64 per stage, LDS row pitch padded to 72
// halves (144B) via the TDM pad feature -> conflict-free ds_load_b128 fragments.
#define KT    64
#define PKT   72
#define PANEL (64 * PKT)

// ---------------- WMMA fragment builders (wave32 layouts, ISA 7.12.2) ----------------
// A 16x32 f16: lane holds row M=lane&15; elems 0-7 <-> K=8*hsel+0..7, 8-15 <-> K=16+8*hsel+0..7
__device__ __forceinline__ v16h frag_a(const _Float16* p, int hsel) {
  v8h lo = *(const v8h*)(p + hsel * 8);
  v8h hi = *(const v8h*)(p + 16 + hsel * 8);
  v16h r;
#pragma unroll
  for (int i = 0; i < 8; ++i) { r[i] = lo[i]; r[i + 8] = hi[i]; }
  return r;
}
// B 32x16: lane holds col N=lane&15; elems 0-7 <-> K=8*hsel+0..7, 8-15 <-> K=16+8*hsel+0..7
__device__ __forceinline__ v16h frag_b(const _Float16* p, int hsel) {
  v8h lo = *(const v8h*)(p + hsel * 8);
  v8h hi = *(const v8h*)(p + 16 + hsel * 8);
  v16h r;
#pragma unroll
  for (int i = 0; i < 8; ++i) { r[i] = lo[i]; r[i + 8] = hi[i]; }
  return r;
}

#if HAVE_TDM
// One 64(rows) x 64(K) f16 tile, row stride `stride_elems`, into LDS at byte
// offset lds_byte with 16B padding per 128B row (pitch 144B).  D# per ISA 8.3/8.4.
__device__ __forceinline__ void tdm_load_tile64x64(unsigned lds_byte,
                                                   const _Float16* g,
                                                   unsigned stride_elems) {
  unsigned long long ga = (unsigned long long)(size_t)g;
  v4u g0;
  g0[0] = 1u;                                             // count=1, no gather
  g0[1] = lds_byte;                                       // lds_addr (bytes)
  g0[2] = (unsigned)ga;                                   // global_addr[31:0]
  g0[3] = (unsigned)((ga >> 32) & 0x01ffffffu) | (2u << 30); // addr[56:32] | type=2
  v8i g1;
  g1[0] = (int)((1u << 16) | (1u << 20) | (4u << 22) | (3u << 25));
          // data_size=2B | pad_enable | pad_interval=32 DW | pad_amount=4 DW
  g1[1] = (int)(64u << 16);        // tensor_dim0[15:0]=64
  g1[2] = (int)(64u << 16);        // tensor_dim0 hi=0 | tensor_dim1[15:0]=64
  g1[3] = (int)(64u << 16);        // tensor_dim1 hi=0 | tile_dim0=64
  g1[4] = (int)64u;                // tile_dim1=64, tile_dim2=0
  g1[5] = (int)stride_elems;       // tensor_dim0_stride[31:0]
  g1[6] = 0;
  g1[7] = 0;
  v4i z = {0, 0, 0, 0};
#if __clang_major__ >= 23
  v8i z8 = {0, 0, 0, 0, 0, 0, 0, 0};
  __builtin_amdgcn_tensor_load_to_lds(g0, g1, z, z, z8, 0);
#else
  __builtin_amdgcn_tensor_load_to_lds(g0, g1, z, z, 0);
#endif
}
#endif

// ---------------- TDM-staged, double-buffered WMMA GEMM ----------------
// C = epi(A @ Bt^T + bias).  128 threads = 4 waves; block tile 64x64, each wave
// a 32x32 (2x2 of 16x16) register tile; K staged through LDS in KT=64 panels.
template<int EPI>
__global__ void __launch_bounds__(128) gemm_wmma_k(
    const _Float16* __restrict__ Ap, const _Float16* __restrict__ Btp,
    void* __restrict__ Cp, _Float16* __restrict__ C2p,
    const float* __restrict__ bias,
    int K, int lda, int ldb, int ldc,
    long long sA, long long sB, long long sC, float scale)
{
  __shared__ _Float16 sm[4 * PANEL];          // {A,B} x double-buffer, 36 KB

  const int tid  = threadIdx.x;
  const int lane = tid & 31;
  const int wave = tid >> 5;
  const int wm   = wave & 1;
  const int wn   = wave >> 1;
  const int hsel = lane >> 4;
  const int l16  = lane & 15;
  const int m0   = blockIdx.x * 64 + wm * 32;
  const int n0   = blockIdx.y * 64 + wn * 32;
  const int bm   = blockIdx.x * 64;           // block-tile origins
  const int bn   = blockIdx.y * 64;

  const _Float16* Ag = Ap  + (size_t)blockIdx.z * sA;
  const _Float16* Bg = Btp + (size_t)blockIdx.z * sB;

#if HAVE_TDM
  const unsigned lds_base = (unsigned)(size_t)(const void*)sm;
#endif

  auto issue_stage = [&](int buf, int k0) {
#if HAVE_TDM
    if (wave == 0) {
      tdm_load_tile64x64(lds_base + (unsigned)(buf * 2 * PANEL) * 2u,
                         Ag + (size_t)bm * lda + k0, (unsigned)lda);
      tdm_load_tile64x64(lds_base + (unsigned)((buf * 2 + 1) * PANEL) * 2u,
                         Bg + (size_t)bn * ldb + k0, (unsigned)ldb);
    }
#else
    _Float16* dA = sm + buf * 2 * PANEL;
    _Float16* dB = dA + PANEL;
    for (int c = tid; c < 64 * 8; c += 128) {       // 64 rows x 8 chunks of 16B
      int row = c >> 3, ch = (c & 7) * 8;
      *(v8h*)(dA + row * PKT + ch) = *(const v8h*)(Ag + (size_t)(bm + row) * lda + k0 + ch);
      *(v8h*)(dB + row * PKT + ch) = *(const v8h*)(Bg + (size_t)(bn + row) * ldb + k0 + ch);
    }
#endif
  };
  auto wait_stage = [&](bool last) {
#if HAVE_TDM
    if (wave == 0) {
      if (last) __builtin_amdgcn_s_wait_tensorcnt(0);   // everything landed
      else      __builtin_amdgcn_s_wait_tensorcnt(2);   // current stage landed
    }
#endif
  };

  const v8f vzero = {0.f, 0.f, 0.f, 0.f, 0.f, 0.f, 0.f, 0.f};
  v8f acc[2][2];
#pragma unroll
  for (int i = 0; i < 2; ++i)
#pragma unroll
    for (int j = 0; j < 2; ++j) acc[i][j] = vzero;

  const int nStages = K / KT;
  issue_stage(0, 0);                                    // prologue DMA

  int buf = 0;
  for (int s = 0; s < nStages; ++s) {
    if (s + 1 < nStages) issue_stage(buf ^ 1, (s + 1) * KT);  // prefetch next panel
    wait_stage(s + 1 >= nStages);
    __syncthreads();                                    // panel visible to all waves

    const _Float16* As = sm + buf * 2 * PANEL;
    const _Float16* Bs = As + PANEL;
#pragma unroll
    for (int ks = 0; ks < KT; ks += 32) {
      v16h a0 = frag_a(As + (wm * 32 + l16)      * PKT + ks, hsel);
      v16h a1 = frag_a(As + (wm * 32 + 16 + l16) * PKT + ks, hsel);
      v16h b0 = frag_b(Bs + (wn * 32 + l16)      * PKT + ks, hsel);
      v16h b1 = frag_b(Bs + (wn * 32 + 16 + l16) * PKT + ks, hsel);
      acc[0][0] = __builtin_amdgcn_wmma_f32_16x16x32_f16(false, a0, false, b0, (short)0, acc[0][0], false, false);
      acc[0][1] = __builtin_amdgcn_wmma_f32_16x16x32_f16(false, a0, false, b1, (short)0, acc[0][1], false, false);
      acc[1][0] = __builtin_amdgcn_wmma_f32_16x16x32_f16(false, a1, false, b0, (short)0, acc[1][0], false, false);
      acc[1][1] = __builtin_amdgcn_wmma_f32_16x16x32_f16(false, a1, false, b1, (short)0, acc[1][1], false, false);
    }
    __syncthreads();                                    // done reading before overwrite
    buf ^= 1;
  }

  // ------- epilogues (D layout: lane col = n&15, VGPR r -> row = r + 8*hsel) -------
#pragma unroll
  for (int ti = 0; ti < 2; ++ti) {
#pragma unroll
    for (int tj = 0; tj < 2; ++tj) {
      const int col   = n0 + tj * 16 + l16;
      const int rbase = m0 + ti * 16 + hsel * 8;
      const v8f c = acc[ti][tj];
      if (EPI == EPI_SIG) {
        float*    O  = (float*)Cp + (size_t)blockIdx.z * sC;
        _Float16* O2 = C2p + (size_t)blockIdx.z * sC;
#pragma unroll
        for (int r = 0; r < 8; ++r) {
          float v = 1.0f / (1.0f + __expf(-c[r] * scale));
          O [(size_t)(rbase + r) * ldc + col] = v;            // f32 attn -> d_out
          O2[(size_t)(rbase + r) * ldc + col] = (_Float16)v;  // f16 copy for attn@V
        }
      } else if (EPI == EPI_F32B) {
        const float bv = bias ? bias[col] : 0.0f;
        float* O = (float*)Cp + (size_t)blockIdx.z * sC;
#pragma unroll
        for (int r = 0; r < 8; ++r)
          O[(size_t)(rbase + r) * ldc + col] = c[r] + bv;
      } else if (EPI == EPI_F16T) {
        // transposed store O[col][row]: rows contiguous per lane -> one 16B store
        const float bv = bias ? bias[col] : 0.0f;
        _Float16* O = (_Float16*)Cp + (size_t)blockIdx.z * sC;
        v8h pk;
#pragma unroll
        for (int r = 0; r < 8; ++r) pk[r] = (_Float16)(c[r] + bv);
        *(v8h*)(O + (size_t)col * ldc + rbase) = pk;
      } else {  // EPI_F16B / EPI_F16RELU
        const float bv = bias ? bias[col] : 0.0f;
        _Float16* O = (_Float16*)Cp + (size_t)blockIdx.z * sC;
#pragma unroll
        for (int r = 0; r < 8; ++r) {
          float v = c[r] + bv;
          if (EPI == EPI_F16RELU) v = fmaxf(v, 0.0f);
          O[(size_t)(rbase + r) * ldc + col] = (_Float16)v;
        }
      }
    }
  }
}

// ---------------- elementwise helpers ----------------
__global__ void addcvt_k(const float* __restrict__ a, const float* __restrict__ b,
                         _Float16* __restrict__ o, int n) {
  int i = blockIdx.x * blockDim.x + threadIdx.x;
  if (i < n) o[i] = (_Float16)(a[i] + b[i]);
}
__global__ void cvt_k(const float* __restrict__ a, _Float16* __restrict__ o, int n) {
  int i = blockIdx.x * blockDim.x + threadIdx.x;
  if (i < n) o[i] = (_Float16)a[i];
}

// ---------------- LayerNorm (one row of E per block) ----------------
template<bool OUT_F16>
__global__ void __launch_bounds__(256) layernorm_k(
    const float* __restrict__ x, const float* __restrict__ g,
    const float* __restrict__ be, void* __restrict__ out, int E)
{
  const int row = blockIdx.x;
  const float* xr = x + (size_t)row * E;
  float s = 0.0f, ss = 0.0f;
  for (int i = threadIdx.x; i < E; i += blockDim.x) {
    float v = xr[i]; s += v; ss += v * v;
  }
  __shared__ float shs[8], shq[8];
  for (int off = 16; off > 0; off >>= 1) {
    s  += __shfl_down(s, off, 32);
    ss += __shfl_down(ss, off, 32);
  }
  const int wid = threadIdx.x >> 5, lid = threadIdx.x & 31;
  if (lid == 0) { shs[wid] = s; shq[wid] = ss; }
  __syncthreads();
  if (wid == 0) {
    s  = (lid < 8) ? shs[lid] : 0.0f;
    ss = (lid < 8) ? shq[lid] : 0.0f;
    for (int off = 4; off > 0; off >>= 1) {
      s  += __shfl_down(s, off, 32);
      ss += __shfl_down(ss, off, 32);
    }
    if (lid == 0) { shs[0] = s; shq[0] = ss; }
  }
  __syncthreads();
  const float mean = shs[0] / (float)E;
  const float var  = shq[0] / (float)E - mean * mean;
  const float inv  = rsqrtf(var + 1e-5f);
  for (int i = threadIdx.x; i < E; i += blockDim.x) {
    float v = (xr[i] - mean) * inv * g[i] + be[i];
    if (OUT_F16) ((_Float16*)out)[(size_t)row * E + i] = (_Float16)v;
    else         ((float*)out)[(size_t)row * E + i] = v;
  }
}

// ---------------- orchestration ----------------
extern "C" void kernel_launch(void* const* d_in, const int* in_sizes, int n_in,
                              void* d_out, int out_size, void* d_ws, size_t ws_size,
                              hipStream_t stream) {
  (void)in_sizes; (void)n_in; (void)out_size; (void)ws_size;
  const float* value  = (const float*)d_in[0];
  const float* key_t  = (const float*)d_in[1];
  const float* query  = (const float*)d_in[2];
  const float* embed0 = (const float*)d_in[3];
  const float* Wq = (const float*)d_in[4];  const float* bq = (const float*)d_in[5];
  const float* Wk = (const float*)d_in[6];  const float* bk = (const float*)d_in[7];
  const float* Wv = (const float*)d_in[8];  const float* bv = (const float*)d_in[9];
  const float* W1 = (const float*)d_in[10]; const float* b1 = (const float*)d_in[11];
  const float* W2 = (const float*)d_in[12]; const float* b2 = (const float*)d_in[13];
  const float* g1 = (const float*)d_in[14]; const float* be1 = (const float*)d_in[15];
  const float* g2 = (const float*)d_in[16]; const float* be2 = (const float*)d_in[17];

  const int S = 4096, E = 1024, F = 4096;
  const int M = 2 * S;                       // 8192 tokens total
  const size_t nTok = (size_t)M * E;

  char* ws = (char*)d_ws; size_t off = 0;
  auto carveH = [&](size_t n) { _Float16* p = (_Float16*)(ws + off); off += n * 2; return p; };
  auto carveF = [&](size_t n) { float*    p = (float*)(ws + off);    off += n * 4; return p; };

  _Float16* qin_h  = carveH(nTok);
  _Float16* kin_h  = carveH(nTok);
  _Float16* vin_h  = carveH(nTok);
  _Float16* Wq_h   = carveH((size_t)E * E);
  _Float16* Wk_h   = carveH((size_t)E * E);
  _Float16* Wv_h   = carveH((size_t)E * E);
  _Float16* W1_h   = carveH((size_t)F * E);
  _Float16* W2_h   = carveH((size_t)E * F);
  _Float16* Qh     = carveH(nTok);
  _Float16* Kh     = carveH(nTok);
  _Float16* Vt     = carveH((size_t)2 * E * S);   // per batch [E][S]
  _Float16* attn_h = carveH((size_t)2 * S * S);   // f16 attn for attn@V
  _Float16* xln_h  = carveH(nTok);
  _Float16* h_h    = carveH((size_t)M * F);
  float*    x_f    = carveF(nTok);
  float*    y_f    = carveF(nTok);

  float* out_f  = (float*)d_out;
  float* attn_f = out_f + nTok;                   // [B][S][S] slot of d_out

  const int EW = 256;
  addcvt_k<<<(int)(nTok / EW), EW, 0, stream>>>(query, embed0, qin_h, (int)nTok);
  addcvt_k<<<(int)(nTok / EW), EW, 0, stream>>>(key_t, embed0, kin_h, (int)nTok);
  addcvt_k<<<(int)(nTok / EW), EW, 0, stream>>>(value, embed0, vin_h, (int)nTok);
  cvt_k<<<(E * E) / EW, EW, 0, stream>>>(Wq, Wq_h, E * E);
  cvt_k<<<(E * E) / EW, EW, 0, stream>>>(Wk, Wk_h, E * E);
  cvt_k<<<(E * E) / EW, EW, 0, stream>>>(Wv, Wv_h, E * E);
  cvt_k<<<(F * E) / EW, EW, 0, stream>>>(W1, W1_h, F * E);
  cvt_k<<<(E * F) / EW, EW, 0, stream>>>(W2, W2_h, E * F);

  const dim3 blk(128);
  // Q = qin @ Wq.T + bq          (M=8192, N=1024, K=1024)
  gemm_wmma_k<EPI_F16B><<<dim3(M / 64, E / 64, 1), blk, 0, stream>>>(
      qin_h, Wq_h, Qh, nullptr, bq, E, E, E, E, 0, 0, 0, 1.0f);
  // K = kin @ Wk.T + bk
  gemm_wmma_k<EPI_F16B><<<dim3(M / 64, E / 64, 1), blk, 0, stream>>>(
      kin_h, Wk_h, Kh, nullptr, bk, E, E, E, E, 0, 0, 0, 1.0f);
  // V, stored transposed per batch: Vt[b][e][s]
  gemm_wmma_k<EPI_F16T><<<dim3(S / 64, E / 64, 2), blk, 0, stream>>>(
      vin_h, Wv_h, Vt, nullptr, bv, E, E, E, /*ldT=*/S,
      (long long)S * E, 0, (long long)E * S, 1.0f);
  // attn = sigmoid(Q K^T / 32): f32 to d_out, f16 copy for next GEMM
  gemm_wmma_k<EPI_SIG><<<dim3(S / 64, S / 64, 2), blk, 0, stream>>>(
      Qh, Kh, attn_f, attn_h, nullptr, E, E, E, S,
      (long long)S * E, (long long)S * E, (long long)S * S, 0.03125f);
  // x = attn @ V                 (M=4096/batch, N=1024, K=4096)
  gemm_wmma_k<EPI_F32B><<<dim3(S / 64, E / 64, 2), blk, 0, stream>>>(
      attn_h, Vt, x_f, nullptr, nullptr, S, S, S, E,
      (long long)S * S, (long long)E * S, (long long)S * E, 1.0f);
  // LN1 -> f16
  layernorm_k<true><<<M, 256, 0, stream>>>(x_f, g1, be1, xln_h, E);
  // h = relu(xln @ W1.T + b1)    (M=8192, N=4096, K=1024)
  gemm_wmma_k<EPI_F16RELU><<<dim3(M / 64, F / 64, 1), blk, 0, stream>>>(
      xln_h, W1_h, h_h, nullptr, b1, E, E, E, F, 0, 0, 0, 1.0f);
  // y = h @ W2.T + b2            (M=8192, N=1024, K=4096)
  gemm_wmma_k<EPI_F32B><<<dim3(M / 64, E / 64, 1), blk, 0, stream>>>(
      h_h, W2_h, y_f, nullptr, b2, F, F, F, E, 0, 0, 0, 1.0f);
  // LN2 -> out (f32)
  layernorm_k<false><<<M, 256, 0, stream>>>(y_f, g2, be2, out_f, E);
}
